// HashingModel_62543313764730
// MI455X (gfx1250) — matrix operand order repaired and checked by reading, not compiled, not run on hardware
//
#include <hip/hip_runtime.h>

// ---------------------------------------------------------------------------
// Types for CDNA5 WMMA (wave32)
// ---------------------------------------------------------------------------
typedef __attribute__((ext_vector_type(8)))  float  v8f;
typedef __attribute__((ext_vector_type(16))) __bf16 v16bf;
typedef __attribute__((ext_vector_type(8)))  __bf16 bf16x8;
typedef __attribute__((ext_vector_type(4)))  __bf16 bf16x4;

#define WMMA_BF16(a, b, c) \
  __builtin_amdgcn_wmma_f32_16x16x32_bf16(false, (a), false, (b), (short)0, (c), false, false)

// Build v16bf from two 8-element bf16 chunks (A-frag: K {0..7} + {16..23} pattern)
__device__ __forceinline__ v16bf frag_cat(bf16x8 lo, bf16x8 hi) {
  v16bf r;
#pragma unroll
  for (int i = 0; i < 8; ++i) { r[i] = lo[i]; r[i + 8] = hi[i]; }
  return r;
}

// Convert 8+8 f32 (two chunks) -> v16bf
__device__ __forceinline__ v16bf cvt8x2(const float* lo, const float* hi) {
  float4 a = *(const float4*)lo;
  float4 b = *(const float4*)(lo + 4);
  float4 c = *(const float4*)hi;
  float4 d = *(const float4*)(hi + 4);
  v16bf r;
  r[0]  = (__bf16)a.x; r[1]  = (__bf16)a.y; r[2]  = (__bf16)a.z; r[3]  = (__bf16)a.w;
  r[4]  = (__bf16)b.x; r[5]  = (__bf16)b.y; r[6]  = (__bf16)b.z; r[7]  = (__bf16)b.w;
  r[8]  = (__bf16)c.x; r[9]  = (__bf16)c.y; r[10] = (__bf16)c.z; r[11] = (__bf16)c.w;
  r[12] = (__bf16)d.x; r[13] = (__bf16)d.y; r[14] = (__bf16)d.z; r[15] = (__bf16)d.w;
  return r;
}
// 16 consecutive f32 -> v16bf
__device__ __forceinline__ v16bf cvt16f(const float* p) { return cvt8x2(p, p + 8); }

__device__ __forceinline__ void st_bf4(__bf16* d, float4 f) {
  bf16x4 h;
  h[0] = (__bf16)f.x; h[1] = (__bf16)f.y; h[2] = (__bf16)f.z; h[3] = (__bf16)f.w;
  *(bf16x4*)d = h;
}

// ---------------------------------------------------------------------------
// CDNA5 async global->LDS copy (16B per lane, tracked by ASYNCcnt).
// ---------------------------------------------------------------------------
__device__ __forceinline__ void async_ld_f4(const float* gsrc, float* lds_dst) {
  unsigned ldso = (unsigned)(unsigned long long)lds_dst;
  asm volatile("global_load_async_to_lds_b128 %0, %1, off"
               :: "v"(ldso), "v"(gsrc)
               : "memory");
}
__device__ __forceinline__ void wait_async8() {
  asm volatile("s_wait_asynccnt 0x8" ::: "memory");
}
__device__ __forceinline__ void wait_async0() {
  asm volatile("s_wait_asynccnt 0x0" ::: "memory");
}

// ---------------------------------------------------------------------------
// Main GEMM: C[M,N] = act(A[M,K] @ W[N,K]^T + bias[N])
// Block tile 128x128, BK=32, 256 threads; 8 waves in 4(M)x2(N), 32x64 each.
// Async DMA global->LDS (f32), double-buffered; f32->bf16 on fragment build.
// Requires: M % 128 == 0, N % 128 == 0, K % 64 == 0 (even # of 32-K steps).
// ---------------------------------------------------------------------------
#define GBK 32
#define TLD 36  // f32 row stride in LDS dwords -> conflict-free b128 frag reads

__device__ __forceinline__ void stage_tiles_async(const float* __restrict__ A, int lda, int bM,
                                                  const float* __restrict__ W, int ldb, int bN,
                                                  int k0, float* Abuf, float* Bbuf, int sidx) {
#pragma unroll
  for (int j = 0; j < 4; ++j) {
    int idx = j * 256 + sidx;
    int r = idx >> 3, c4 = (idx & 7) * 4;
    async_ld_f4(A + (bM + r) * lda + k0 + c4, Abuf + r * TLD + c4);
  }
#pragma unroll
  for (int j = 0; j < 4; ++j) {
    int idx = j * 256 + sidx;
    int r = idx >> 3, c4 = (idx & 7) * 4;
    async_ld_f4(W + (bN + r) * ldb + k0 + c4, Bbuf + r * TLD + c4);
  }
}

// One 32-K step: load ALL fragments first (distinct regs), then 8 back-to-back
// WMMAs (no WAR hazard NOPs, ds waits overlap across fragments).
__device__ __forceinline__ void gemm_step(const float* __restrict__ Ab,
                                          const float* __restrict__ Bb,
                                          int wm, int wn, int l15, int kh,
                                          v8f acc0[4], v8f acc1[4]) {
  const float* pa0 = Ab + (wm * 32 + l15) * TLD + kh * 8;
  const float* pa1 = Ab + (wm * 32 + 16 + l15) * TLD + kh * 8;
  v16bf a0 = cvt8x2(pa0, pa0 + 16);
  v16bf a1 = cvt8x2(pa1, pa1 + 16);
  v16bf bfr[4];
#pragma unroll
  for (int nt = 0; nt < 4; ++nt) {
    const float* pb = Bb + (wn * 64 + nt * 16 + l15) * TLD + kh * 8;
    bfr[nt] = cvt8x2(pb, pb + 16);
  }
#pragma unroll
  for (int nt = 0; nt < 4; ++nt) acc0[nt] = WMMA_BF16(a0, bfr[nt], acc0[nt]);
#pragma unroll
  for (int nt = 0; nt < 4; ++nt) acc1[nt] = WMMA_BF16(a1, bfr[nt], acc1[nt]);
}

__launch_bounds__(256)
__global__ void gemm_bf16_async(const float* __restrict__ A, int lda,
                                const float* __restrict__ W, int ldb,
                                const float* __restrict__ bias,
                                float* __restrict__ C, int ldc,
                                int K, int relu) {
  __shared__ float Asf[2][128 * TLD];
  __shared__ float Bsf[2][128 * TLD];
  const int tid  = threadIdx.x;
  const int lane = tid & 31, w = tid >> 5;
  const int wm = w >> 1, wn = w & 1;      // 4 x 2 waves, 32x64 tiles
  const int bM = blockIdx.x * 128;
  const int bN = blockIdx.y * 128;
  const int l15 = lane & 15, kh = lane >> 4;
  const int sidx = (w * 32 + lane);

  v8f acc0[4] = {{}, {}, {}, {}};
  v8f acc1[4] = {{}, {}, {}, {}};

  // prologue: stage k-tile 0 into buffer 0
  stage_tiles_async(A, lda, bM, W, ldb, bN, 0, Asf[0], Bsf[0], sidx);

  const int nk = K / GBK;  // even for all shapes used here
  for (int i = 0; i < nk; i += 2) {
    // ---- even step: compute buf0, prefetch into buf1 ----
    if (i + 1 < nk) {
      stage_tiles_async(A, lda, bM, W, ldb, bN, (i + 1) * GBK, Asf[1], Bsf[1], sidx);
      wait_async8();       // in-order completion: buf0's copies done
    } else {
      wait_async0();
    }
    __syncthreads();
    gemm_step(Asf[0], Bsf[0], wm, wn, l15, kh, acc0, acc1);
    __syncthreads();

    if (i + 1 < nk) {
      // ---- odd step: compute buf1, prefetch into buf0 ----
      if (i + 2 < nk) {
        stage_tiles_async(A, lda, bM, W, ldb, bN, (i + 2) * GBK, Asf[0], Bsf[0], sidx);
        wait_async8();
      } else {
        wait_async0();
      }
      __syncthreads();
      gemm_step(Asf[1], Bsf[1], wm, wn, l15, kh, acc0, acc1);
      __syncthreads();
    }
  }

  // Epilogue: C layout lane<16 -> M=v, lane>=16 -> M=v+8; N = lane&15
  const int mb = bM + wm * 32 + 8 * kh;
  const int nb = bN + wn * 64 + l15;
#pragma unroll
  for (int nt = 0; nt < 4; ++nt) {
    int n = nb + nt * 16;
    float bv = bias[n];
#pragma unroll
    for (int v = 0; v < 8; ++v) {
      float x = acc0[nt][v] + bv;
      if (relu) x = fmaxf(x, 0.0f);
      C[(mb + v) * ldc + n] = x;
    }
#pragma unroll
    for (int v = 0; v < 8; ++v) {
      float x = acc1[nt][v] + bv;
      if (relu) x = fmaxf(x, 0.0f);
      C[(mb + 16 + v) * ldc + n] = x;
    }
  }
}

// ---------------------------------------------------------------------------
// Small-N GEMM (N multiple of 64): block tile 128x64, BK=32, VALU staging.
// Used for the final N=64 hash projection.
// ---------------------------------------------------------------------------
#define ALD 40
#define BLD 40

__launch_bounds__(256)
__global__ void gemm_bf16_small(const float* __restrict__ A, int lda,
                                const float* __restrict__ W, int ldb,
                                const float* __restrict__ bias,
                                float* __restrict__ C, int ldc,
                                int K, int relu) {
  __shared__ __bf16 As[128 * ALD];
  __shared__ __bf16 Bs[64 * BLD];
  const int tid  = threadIdx.x;
  const int lane = tid & 31, w = tid >> 5;
  const int wm = w >> 1, wn = w & 1;
  const int bM = blockIdx.x * 128;
  const int bN = blockIdx.y * 64;
  const int l15 = lane & 15, kh = lane >> 4;

  v8f a00 = {}, a01 = {}, a10 = {}, a11 = {};

  for (int k0 = 0; k0 < K; k0 += GBK) {
#pragma unroll
    for (int i = 0; i < 4; ++i) {
      int idx = tid + i * 256;
      int r = idx >> 3, c4 = (idx & 7) * 4;
      float4 f = *(const float4*)(A + (bM + r) * lda + k0 + c4);
      st_bf4(&As[r * ALD + c4], f);
    }
#pragma unroll
    for (int i = 0; i < 2; ++i) {
      int idx = tid + i * 256;
      int r = idx >> 3, c4 = (idx & 7) * 4;
      float4 f = *(const float4*)(W + (bN + r) * ldb + k0 + c4);
      st_bf4(&Bs[r * BLD + c4], f);
    }
    __syncthreads();

    const __bf16* ar0 = &As[(wm * 32 + l15) * ALD + kh * 8];
    const __bf16* ar1 = &As[(wm * 32 + 16 + l15) * ALD + kh * 8];
    v16bf af0 = frag_cat(*(const bf16x8*)ar0, *(const bf16x8*)(ar0 + 16));
    v16bf af1 = frag_cat(*(const bf16x8*)ar1, *(const bf16x8*)(ar1 + 16));
    const __bf16* br0 = &Bs[(wn * 32 + l15) * BLD + kh * 8];
    const __bf16* br1 = &Bs[(wn * 32 + 16 + l15) * BLD + kh * 8];
    v16bf bf0 = frag_cat(*(const bf16x8*)br0, *(const bf16x8*)(br0 + 16));
    v16bf bf1 = frag_cat(*(const bf16x8*)br1, *(const bf16x8*)(br1 + 16));

    a00 = WMMA_BF16(af0, bf0, a00);
    a01 = WMMA_BF16(af0, bf1, a01);
    a10 = WMMA_BF16(af1, bf0, a10);
    a11 = WMMA_BF16(af1, bf1, a11);
    __syncthreads();
  }

  const int mb = bM + wm * 32 + 8 * kh;
  const int nb = bN + wn * 32 + l15;
#pragma unroll
  for (int t = 0; t < 4; ++t) {
    const v8f* acc = (t == 0) ? &a00 : (t == 1) ? &a01 : (t == 2) ? &a10 : &a11;
    int m0 = mb + ((t >> 1) ? 16 : 0);
    int n  = nb + ((t & 1) ? 16 : 0);
    float bv = bias[n];
#pragma unroll
    for (int v = 0; v < 8; ++v) {
      float x = (*acc)[v] + bv;
      if (relu) x = fmaxf(x, 0.0f);
      C[(m0 + v) * ldc + n] = x;
    }
  }
}

// ---------------------------------------------------------------------------
// Prompt scoring + argmax + gather (exact fp32).
// ---------------------------------------------------------------------------
__launch_bounds__(256)
__global__ void argmax_gather(const float* __restrict__ X,
                              const float* __restrict__ Pm,
                              float* __restrict__ rm) {
  __shared__ float Xs[16][512];
  __shared__ float wbv[8][16];
  __shared__ int   wbi[8][16];
  __shared__ int   fbi[16];
  const int tid = threadIdx.x, lane = tid & 31, w = tid >> 5;
  const int bM = blockIdx.x * 16;

#pragma unroll
  for (int i = 0; i < 8; ++i) {
    int idx = tid + i * 256;
    int r = idx >> 7, c4 = (idx & 127) * 4;
    *(float4*)&Xs[r][c4] = *(const float4*)(X + (bM + r) * 512 + c4);
  }
  __syncthreads();

  float bestv[16];
  int   besti[16];
#pragma unroll
  for (int r = 0; r < 16; ++r) { bestv[r] = -3.4e38f; besti[r] = 0x7fffffff; }

  for (int g = 0; g < 4; ++g) {
    int p0 = g * 1024 + tid;
    const float4* q0 = (const float4*)(Pm + (p0      ) * 512);
    const float4* q1 = (const float4*)(Pm + (p0 + 256) * 512);
    const float4* q2 = (const float4*)(Pm + (p0 + 512) * 512);
    const float4* q3 = (const float4*)(Pm + (p0 + 768) * 512);
    float acc[4][16];
#pragma unroll
    for (int j = 0; j < 4; ++j)
#pragma unroll
      for (int r = 0; r < 16; ++r) acc[j][r] = 0.0f;

    for (int k4 = 0; k4 < 128; ++k4) {
      float4 f0 = q0[k4], f1 = q1[k4], f2 = q2[k4], f3 = q3[k4];
#pragma unroll
      for (int r = 0; r < 16; ++r) {
        float x0 = Xs[r][k4 * 4 + 0], x1 = Xs[r][k4 * 4 + 1];
        float x2 = Xs[r][k4 * 4 + 2], x3 = Xs[r][k4 * 4 + 3];
        acc[0][r] += f0.x * x0 + f0.y * x1 + f0.z * x2 + f0.w * x3;
        acc[1][r] += f1.x * x0 + f1.y * x1 + f1.z * x2 + f1.w * x3;
        acc[2][r] += f2.x * x0 + f2.y * x1 + f2.z * x2 + f2.w * x3;
        acc[3][r] += f3.x * x0 + f3.y * x1 + f3.z * x2 + f3.w * x3;
      }
    }
#pragma unroll
    for (int j = 0; j < 4; ++j) {
      int p = p0 + j * 256;
#pragma unroll
      for (int r = 0; r < 16; ++r) {
        float v = acc[j][r];
        if (v > bestv[r] || (v == bestv[r] && p < besti[r])) { bestv[r] = v; besti[r] = p; }
      }
    }
  }

#pragma unroll
  for (int r = 0; r < 16; ++r) {
    float v = bestv[r]; int i = besti[r];
    for (int off = 16; off; off >>= 1) {
      float ov = __shfl_xor(v, off, 32);
      int   oi = __shfl_xor(i, off, 32);
      if (ov > v || (ov == v && oi < i)) { v = ov; i = oi; }
    }
    if (lane == 0) { wbv[w][r] = v; wbi[w][r] = i; }
  }
  __syncthreads();
  if (tid < 16) {
    float v = wbv[0][tid]; int i = wbi[0][tid];
    for (int ww = 1; ww < 8; ++ww) {
      float ov = wbv[ww][tid]; int oi = wbi[ww][tid];
      if (ov > v || (ov == v && oi < i)) { v = ov; i = oi; }
    }
    fbi[tid] = i;
  }
  __syncthreads();
#pragma unroll
  for (int r = 0; r < 16; ++r) {
    const float2* src = (const float2*)(Pm + (long)fbi[r] * 512);
    float2* dst = (float2*)(rm + (bM + r) * 512);
    dst[tid] = src[tid];
  }
}

// ---------------------------------------------------------------------------
// V transpose: out[c*2048 + r] = in[r*ldin + c]   (2048 x 512)
// ---------------------------------------------------------------------------
__global__ void transpose_v(const float* __restrict__ in, int ldin, float* __restrict__ out) {
  __shared__ float t[32][33];
  int r0 = blockIdx.x * 32, c0 = blockIdx.y * 32;
  int tx = threadIdx.x, ty = threadIdx.y;
#pragma unroll
  for (int j = 0; j < 4; ++j) t[ty + j * 8][tx] = in[(r0 + ty + j * 8) * ldin + c0 + tx];
  __syncthreads();
#pragma unroll
  for (int j = 0; j < 4; ++j) out[(c0 + ty + j * 8) * 2048 + r0 + tx] = t[tx][ty + j * 8];
}

// ---------------------------------------------------------------------------
// Attention: one block per (16-query tile, head). S=2048 keys, HD=64.
// ---------------------------------------------------------------------------
#define PSTR 2056  // bf16 stride: 1028 dwords -> conflict-free b128 frag reads

__launch_bounds__(256)
__global__ void attn_kernel(const float* __restrict__ qp, int ldq,
                            const float* __restrict__ kp, int ldk,
                            const float* __restrict__ vT,
                            float* __restrict__ attO, int ldo) {
  __shared__ __bf16 Ps[16 * PSTR];
  __shared__ float  Op[8 * 256];
  __shared__ float  redbuf[256];
  __shared__ float  rmaxs[16];
  __shared__ float  rsumi[16];

  const int qb = blockIdx.x, h = blockIdx.y;
  const int tid = threadIdx.x, lane = tid & 31, w = tid >> 5;
  const int l15 = lane & 15, kh = lane >> 4;

  const float* qrow = qp + (qb * 16 + l15) * ldq + h * 64;
  v16bf qf0 = cvt8x2(qrow + kh * 8,      qrow + kh * 8 + 16);
  v16bf qf1 = cvt8x2(qrow + 32 + kh * 8, qrow + 32 + kh * 8 + 16);

  for (int t = w; t < 128; t += 8) {
    int s0 = t * 16;
    const float* krow = kp + (s0 + l15) * ldk + h * 64;
    v16bf kf0 = cvt8x2(krow + kh * 8, krow + kh * 8 + 16);
    v16bf kf1 = cvt8x2(krow + 32 + kh * 8, krow + 32 + kh * 8 + 16);
    v8f s = {};
    s = WMMA_BF16(qf0, kf0, s);
    s = WMMA_BF16(qf1, kf1, s);
#pragma unroll
    for (int v = 0; v < 8; ++v)
      Ps[(kh * 8 + v) * PSTR + s0 + l15] = (__bf16)(s[v] * 0.125f);
  }
  __syncthreads();

  {
    int r = tid >> 4, sub = tid & 15;
    __bf16* prow = Ps + r * PSTR + sub * 128;
    float mx = -3.4e38f;
    for (int c = 0; c < 128; ++c) mx = fmaxf(mx, (float)prow[c]);
    redbuf[r * 16 + sub] = mx;
    __syncthreads();
    if (sub == 0) {
      float m = redbuf[r * 16];
      for (int j = 1; j < 16; ++j) m = fmaxf(m, redbuf[r * 16 + j]);
      rmaxs[r] = m;
    }
    __syncthreads();
    float m = rmaxs[r], ssum = 0.0f;
    for (int c = 0; c < 128; ++c) {
      float e = __expf((float)prow[c] - m);
      prow[c] = (__bf16)e;
      ssum += e;
    }
    redbuf[r * 16 + sub] = ssum;
    __syncthreads();
    if (sub == 0) {
      float sm = 0.0f;
      for (int j = 0; j < 16; ++j) sm += redbuf[r * 16 + j];
      rsumi[r] = 1.0f / sm;
    }
  }
  __syncthreads();

  {
    int nt = w & 3, half = w >> 2;
    v8f o = {};
    const __bf16* prow = Ps + l15 * PSTR + kh * 8;
    const float*  vrow = vT + (h * 64 + nt * 16 + l15) * 2048 + kh * 8;
    for (int kt = half * 32; kt < half * 32 + 32; ++kt) {
      int k0 = kt * 32;
      v16bf af = frag_cat(*(const bf16x8*)(prow + k0), *(const bf16x8*)(prow + k0 + 16));
      v16bf bf = cvt8x2(vrow + k0, vrow + k0 + 16);
      o = WMMA_BF16(af, bf, o);
    }
#pragma unroll
    for (int v = 0; v < 8; ++v)
      Op[w * 256 + (kh * 8 + v) * 16 + l15] = o[v];
  }
  __syncthreads();

#pragma unroll
  for (int i = 0; i < 4; ++i) {
    int e = tid + i * 256;
    int mm = e >> 6, c = e & 63;
    int nt = c >> 4, nn = c & 15;
    float val = Op[nt * 256 + mm * 16 + nn] + Op[(nt + 4) * 256 + mm * 16 + nn];
    attO[(qb * 16 + mm) * ldo + h * 64 + c] = val * rsumi[mm];
  }
}

// ---------------------------------------------------------------------------
// Copy X (2048x512, ld 512) into dst (ld 1024) first half
// ---------------------------------------------------------------------------
__global__ void copy_cols512(float* __restrict__ dst, int ldd,
                             const float* __restrict__ src, int lds_) {
  int idx = blockIdx.x * 256 + threadIdx.x;
  int r = idx >> 9, c = idx & 511;
  dst[r * ldd + c] = src[r * lds_ + c];
}

// ---------------------------------------------------------------------------
// Row-wise L2 normalization, 64 cols/row, one wave per row (8 rows/block)
// ---------------------------------------------------------------------------
__launch_bounds__(256)
__global__ void l2norm64(float* __restrict__ Y) {
  int tid = threadIdx.x, lane = tid & 31, w = tid >> 5;
  long row = (long)blockIdx.x * 8 + w;
  float* p = Y + row * 64;
  float x0 = p[lane], x1 = p[lane + 32];
  float s = x0 * x0 + x1 * x1;
#pragma unroll
  for (int off = 16; off; off >>= 1) s += __shfl_xor(s, off, 32);
  float inv = 1.0f / fmaxf(sqrtf(s), 1e-12f);
  p[lane] = x0 * inv;
  p[lane + 32] = x1 * inv;
}

// ---------------------------------------------------------------------------
// Orchestration
// ---------------------------------------------------------------------------
extern "C" void kernel_launch(void* const* d_in, const int* in_sizes, int n_in,
                              void* d_out, int out_size, void* d_ws, size_t ws_size,
                              hipStream_t stream) {
  (void)in_sizes; (void)n_in; (void)out_size; (void)ws_size;

  const float* X[2]     = {(const float*)d_in[0], (const float*)d_in[1]};
  const float* prom     = (const float*)d_in[2];
  const float* in_w[2]  = {(const float*)d_in[3], (const float*)d_in[7]};
  const float* in_b[2]  = {(const float*)d_in[4], (const float*)d_in[8]};
  const float* out_w[2] = {(const float*)d_in[5], (const float*)d_in[9]};
  const float* out_b[2] = {(const float*)d_in[6], (const float*)d_in[10]};
  const float* W1[2] = {(const float*)d_in[11], (const float*)d_in[17]};
  const float* b1[2] = {(const float*)d_in[12], (const float*)d_in[18]};
  const float* W2[2] = {(const float*)d_in[13], (const float*)d_in[19]};
  const float* b2[2] = {(const float*)d_in[14], (const float*)d_in[20]};
  const float* Wc[2] = {(const float*)d_in[15], (const float*)d_in[21]};
  const float* bc[2] = {(const float*)d_in[16], (const float*)d_in[22]};

  float* ws = (float*)d_ws;
  const size_t MF = 1048576;            // 2048*512 floats
  float* rm   = ws;                     // [2048,512]
  float* qp   = ws + 1 * MF;            // [2048,512]
  float* kvp  = ws + 2 * MF;            // [2048,1024]  (k | v)
  float* vT   = ws + 4 * MF;            // [512,2048]
  float* attO = ws + 5 * MF;            // [2048,512]
  float* fi   = ws + 6 * MF;            // [2048,1024]
  float* ft   = ws + 8 * MF;            // [2048,1024]
  float* Y1   = ws + 10 * MF;           // [2048,4096]
  float* Y2   = ws + 18 * MF;           // [2048,4096]  (total 26 MF = 104 MB)

  float* F[2] = {fi, ft};

  for (int m = 0; m < 2; ++m) {
    argmax_gather<<<128, 256, 0, stream>>>(X[m], prom, rm);
    // qp = X @ Wq^T + bq
    gemm_bf16_async<<<dim3(16, 4), 256, 0, stream>>>(X[m], 512, in_w[m], 512, in_b[m],
                                                     qp, 512, 512, 0);
    // kvp = rm @ in_w[512:1536]^T + in_b[512:1536]
    gemm_bf16_async<<<dim3(16, 8), 256, 0, stream>>>(rm, 512, in_w[m] + 512 * 512, 512,
                                                     in_b[m] + 512, kvp, 1024, 512, 0);
    transpose_v<<<dim3(64, 16), dim3(32, 8), 0, stream>>>(kvp + 512, 1024, vT);
    attn_kernel<<<dim3(128, 8), 256, 0, stream>>>(qp, 512, kvp, 1024, vT, attO, 512);
    // f = [X | attO @ out_w^T + out_b]
    copy_cols512<<<4096, 256, 0, stream>>>(F[m], 1024, X[m], 512);
    gemm_bf16_async<<<dim3(16, 4), 256, 0, stream>>>(attO, 512, out_w[m], 512, out_b[m],
                                                     F[m] + 512, 1024, 512, 0);
  }

  float* OUT = (float*)d_out;
  // d_out order: image_hash(fi,img), text_hash(ft,txt), distill_i(ft,img), distill_t(fi,txt)
  const float* Fp[4] = {fi, ft, ft, fi};
  const int    wm4[4] = {0, 1, 0, 1};
  for (int p = 0; p < 4; ++p) {
    int mw = wm4[p];
    gemm_bf16_async<<<dim3(16, 32), 256, 0, stream>>>(Fp[p], 1024, W1[mw], 1024, b1[mw],
                                                      Y1, 4096, 1024, 1);
    gemm_bf16_async<<<dim3(16, 32), 256, 0, stream>>>(Y1, 4096, W2[mw], 4096, b2[mw],
                                                      Y2, 4096, 4096, 1);
    gemm_bf16_small<<<dim3(16, 1), 256, 0, stream>>>(Y2, 4096, Wc[mw], 4096, bc[mw],
                                                     OUT + (size_t)p * 131072, 64, 4096, 0);
  }
  l2norm64<<<1024, 256, 0, stream>>>(OUT);
}
